// MultiHeadAttention_45792941310268
// MI455X (gfx1250) — matrix operand order repaired and checked by reading, compiled, not run on hardware
//
#include <hip/hip_runtime.h>
#include <hip/hip_bf16.h>

// MHA forward for B=2,S=2048,D=1024,H=16,DK=64 on gfx1250 (wave32, WMMA).
// cvt(f32->f16) -> 3 proj GEMMs (WMMA f16, async-LDS double-buffered) ->
// causal flash attention (WMMA f16, f32 online softmax) -> output proj (f32+bias).

typedef __attribute__((ext_vector_type(16))) _Float16 v16h;
typedef __attribute__((ext_vector_type(8)))  _Float16 v8h;
typedef __attribute__((ext_vector_type(8)))  float    v8f;
typedef __attribute__((ext_vector_type(4)))  int      v4i;

#define BB 2
#define SS 2048
#define DD 1024
#define HH 16
#define DKK 64
#define MM (BB*SS)        // 4096 rows
#define KKDIM 1024
#define LDSPITCH 40       // halves per tile row (80B): 20*r mod 64 -> conflict-free b128

#define GLOBAL_AS __attribute__((address_space(1)))
#define LDS_AS    __attribute__((address_space(3)))

#if defined(__AMDGCN__) && __has_builtin(__builtin_amdgcn_global_load_async_to_lds_b128)
#define HAVE_ASYNC_LDS 1
#endif

// ---- fragment loaders (contiguous k = 16*half + s order, same for A and B) -------------------------------
// A 16x32 f16: lane<16 row M=l holds K 0..15 contiguous; lanes 16-31
// row M=l-16 hold K 16..31. p already includes +16*lhalf.
__device__ inline v16h load_frag_a(const _Float16* p) {
  union { v16h v; v8h h[2]; } f;
  f.h[0] = *(const v8h*)(p);
  f.h[1] = *(const v8h*)(p + 8);
  return f.v;
}
// B 32x16 f16: lane<16 column N=l holds K 0..15 contiguous; lanes 16-31 hold
// K 16..31. p already includes +16*lhalf.
__device__ inline v16h load_frag_b(const _Float16* p) {
  union { v16h v; v8h h[2]; } f;
  f.h[0] = *(const v8h*)(p);
  f.h[1] = *(const v8h*)(p + 8);
  return f.v;
}

// ---- async global->LDS 16B copy (per-lane), with safe fallback -----------
__device__ inline void cp16_tile(const _Float16* g, _Float16* l) {
#if defined(HAVE_ASYNC_LDS)
  GLOBAL_AS v4i* gp = (GLOBAL_AS v4i*)(GLOBAL_AS void*)g;
  LDS_AS    v4i* lp = (LDS_AS    v4i*)(LDS_AS    void*)l;
  __builtin_amdgcn_global_load_async_to_lds_b128(gp, lp, 0, 0);
#else
  *(v8h*)l = *(const v8h*)g;
#endif
}

__device__ inline void wait_async_tile() {
#if defined(HAVE_ASYNC_LDS)
#if __has_builtin(__builtin_amdgcn_s_wait_asynccnt)
  __builtin_amdgcn_s_wait_asynccnt(6);   // keep the newest tile (6 ops) in flight
#else
  asm volatile("s_wait_asynccnt 6" ::: "memory");
#endif
#endif
}
__device__ inline void wait_async_all() {
#if defined(HAVE_ASYNC_LDS)
#if __has_builtin(__builtin_amdgcn_s_wait_asynccnt)
  __builtin_amdgcn_s_wait_asynccnt(0);
#else
  asm volatile("s_wait_asynccnt 0" ::: "memory");
#endif
#endif
}

// ---- f32 -> f16 convert --------------------------------------------------
__global__ __launch_bounds__(256) void cvt_f32_f16_kernel(const float* __restrict__ src,
                                                          _Float16* __restrict__ dst, int n) {
  int i = (blockIdx.x * 256 + threadIdx.x) * 8;
  if (i + 8 > n) return;
  float4 f0 = *(const float4*)(src + i);
  float4 f1 = *(const float4*)(src + i + 4);
  v8h h;
  h[0] = (_Float16)f0.x; h[1] = (_Float16)f0.y; h[2] = (_Float16)f0.z; h[3] = (_Float16)f0.w;
  h[4] = (_Float16)f1.x; h[5] = (_Float16)f1.y; h[6] = (_Float16)f1.z; h[7] = (_Float16)f1.w;
  *(v8h*)(dst + i) = h;
}

// ---- GEMM: C[M,N] = A[M,K] @ Bt[N,K]^T + bias ----------------------------
// Double-buffered async LDS staging: WG tile 64x128, 4 waves, each wave 32x64.
// MODE 0: f32 out [m*D+n]; MODE 1: f16 Qh/Kh [B,H,S,64]; MODE 2: f16 Vt [B,H,64,S]
template<int MODE>
__global__ __launch_bounds__(128) void gemm_wmma_kernel(const _Float16* __restrict__ A,
                                                        const _Float16* __restrict__ Bt,
                                                        const float* __restrict__ bias,
                                                        void* __restrict__ outp) {
  __shared__ _Float16 aT[2][64 * LDSPITCH];    // 64 rows x 32 K (padded)
  __shared__ _Float16 bT[2][128 * LDSPITCH];   // 128 N-rows x 32 K (padded)
  const int tid   = threadIdx.x;
  const int lane  = tid & 31;
  const int wave  = tid >> 5;
  const int lrow  = lane & 15;
  const int lhalf = lane >> 4;
  const int rowBase = blockIdx.x * 64;
  const int colBase = blockIdx.y * 128;
  const int wRow = (wave >> 1) * 32;
  const int wCol = (wave & 1)  * 64;

  auto issue_tile = [&](int kk, int buf) {
    // A: 256 16B-chunks, 2/thread.  B: 512 chunks, 4/thread.  6 async ops/wave.
#pragma unroll
    for (int j = 0; j < 2; ++j) {
      const int c = tid + 128 * j;
      const int r = c >> 2, colh = (c & 3) * 8;
      cp16_tile(A + (size_t)(rowBase + r) * KKDIM + kk + colh,
                &aT[buf][r * LDSPITCH + colh]);
    }
#pragma unroll
    for (int j = 0; j < 4; ++j) {
      const int c = tid + 128 * j;
      const int r = c >> 2, colh = (c & 3) * 8;
      cp16_tile(Bt + (size_t)(colBase + r) * KKDIM + kk + colh,
                &bT[buf][r * LDSPITCH + colh]);
    }
  };

  v8f acc[2][4] = {};
  issue_tile(0, 0);

  const int nsteps = KKDIM / 32;
  for (int i = 0; i < nsteps; ++i) {
    const int buf = i & 1;
    if (i + 1 < nsteps) { issue_tile((i + 1) * 32, buf ^ 1); wait_async_tile(); }
    else                { wait_async_all(); }
    __syncthreads();     // whole tile (all waves' portions) visible in LDS

    v16h a[2], b[4];
#pragma unroll
    for (int mi = 0; mi < 2; ++mi)
      a[mi] = load_frag_a(&aT[buf][(wRow + mi * 16 + lrow) * LDSPITCH + 16 * lhalf]);
#pragma unroll
    for (int ni = 0; ni < 4; ++ni)
      b[ni] = load_frag_b(&bT[buf][(wCol + ni * 16 + lrow) * LDSPITCH + 16 * lhalf]);
#pragma unroll
    for (int mi = 0; mi < 2; ++mi)
#pragma unroll
      for (int ni = 0; ni < 4; ++ni)
        acc[mi][ni] = __builtin_amdgcn_wmma_f32_16x16x32_f16(
            false, a[mi], false, b[ni], (short)0, acc[mi][ni], false, false);

    __syncthreads();     // reads done before next iteration's copies land
  }

#pragma unroll
  for (int mi = 0; mi < 2; ++mi) {
#pragma unroll
    for (int ni = 0; ni < 4; ++ni) {
      const int n  = colBase + wCol + ni * 16 + lrow;
      const float bv = bias[n];
#pragma unroll
      for (int r = 0; r < 8; ++r) {
        const int m = rowBase + wRow + mi * 16 + r + 8 * lhalf;
        const float val = acc[mi][ni][r] + bv;
        if (MODE == 0) {
          ((float*)outp)[(size_t)m * DD + n] = val;
        } else {
          const int bb = m >> 11, s = m & (SS - 1);
          const int hh = n >> 6,  dk = n & (DKK - 1);
          _Float16* o = (_Float16*)outp;
          if (MODE == 1)
            o[(((size_t)(bb * HH + hh)) * SS + s) * DKK + dk] = (_Float16)val;
          else
            o[(((size_t)(bb * HH + hh)) * DKK + dk) * SS + s] = (_Float16)val;
        }
      }
    }
  }
}

// ---- causal flash attention: one wave per 16-query tile ------------------
__global__ __launch_bounds__(128) void attn_wmma_kernel(const _Float16* __restrict__ Qh,
                                                        const _Float16* __restrict__ Kh,
                                                        const _Float16* __restrict__ Vt,
                                                        _Float16* __restrict__ att) {
  __shared__ _Float16 pbuf[4][16 * LDSPITCH];  // per-wave P staging, padded
  const int lane  = threadIdx.x & 31;
  const int wave  = threadIdx.x >> 5;
  const int gw    = blockIdx.x * 4 + wave;     // 0..4095
  const int qt    = gw & 127;
  const int bh    = gw >> 7;                   // 0..31
  const int q0    = qt * 16;
  const int lrow  = lane & 15;
  const int lhalf = lane >> 4;
  const size_t baseQK = (size_t)bh * SS * DKK;
  const size_t baseVt = (size_t)bh * DKK * SS;
  _Float16* pb = pbuf[wave];

  const _Float16* qp = Qh + baseQK + (size_t)(q0 + lrow) * DKK + 16 * lhalf;
  v16h aq0 = load_frag_a(qp);
  v16h aq1 = load_frag_a(qp + 32);

  v8f oacc[4] = {};
  float mi[8], li[8];
#pragma unroll
  for (int r = 0; r < 8; ++r) { mi[r] = -1e30f; li[r] = 0.f; }

  for (int j0 = 0; j0 <= q0 + 15; j0 += 32) {
    if (j0 + 32 <= q0 + 15) {   // prefetch next K/V tiles into L2/WGP$
      __builtin_prefetch(Kh + baseQK + (size_t)(j0 + 32 + lrow) * DKK, 0, 0);
      __builtin_prefetch(Vt + baseVt + (size_t)lrow * SS + j0 + 32, 0, 0);
    }
    v8f p[2];
    const bool diag = (j0 + 31 > q0);          // wave-uniform: mask only on diagonal
    // ---- scores: two 16-key subtiles, K-dim = DK = 64 (chained WMMA) ----
#pragma unroll
    for (int t = 0; t < 2; ++t) {
      const int k0 = j0 + t * 16;
      const _Float16* kp = Kh + baseQK + (size_t)(k0 + lrow) * DKK + 16 * lhalf;
      v16h bk0 = load_frag_b(kp);
      v16h bk1 = load_frag_b(kp + 32);
      v8f sc = {};
      sc = __builtin_amdgcn_wmma_f32_16x16x32_f16(false, aq0, false, bk0, (short)0, sc, false, false);
      sc = __builtin_amdgcn_wmma_f32_16x16x32_f16(false, aq1, false, bk1, (short)0, sc, false, false);
      if (diag) {
        const int kcol = k0 + lrow;
#pragma unroll
        for (int r = 0; r < 8; ++r) {
          const int qrow = q0 + r + 8 * lhalf;
          float v = sc[r] * 0.125f;
          if (kcol > qrow) v = -1e30f;
          p[t][r] = v;
        }
      } else {
#pragma unroll
        for (int r = 0; r < 8; ++r) p[t][r] = sc[r] * 0.125f;
      }
    }
    // ---- online softmax (rows span the 16 lanes of each half-wave) ------
    float alpha[8];
#pragma unroll
    for (int r = 0; r < 8; ++r) {
      float v = fmaxf(p[0][r], p[1][r]);
      v = fmaxf(v, __shfl_xor(v, 1, 16));
      v = fmaxf(v, __shfl_xor(v, 2, 16));
      v = fmaxf(v, __shfl_xor(v, 4, 16));
      v = fmaxf(v, __shfl_xor(v, 8, 16));
      const float mn = fmaxf(mi[r], v);
      alpha[r] = __expf(mi[r] - mn);
      mi[r] = mn;
    }
#pragma unroll
    for (int r = 0; r < 8; ++r) {
      p[0][r] = __expf(p[0][r] - mi[r]);
      p[1][r] = __expf(p[1][r] - mi[r]);
      float rs = p[0][r] + p[1][r];
      rs += __shfl_xor(rs, 1, 16);
      rs += __shfl_xor(rs, 2, 16);
      rs += __shfl_xor(rs, 4, 16);
      rs += __shfl_xor(rs, 8, 16);
      li[r] = li[r] * alpha[r] + rs;
    }
#pragma unroll
    for (int ni = 0; ni < 4; ++ni)
#pragma unroll
      for (int r = 0; r < 8; ++r)
        oacc[ni][r] *= alpha[r];
    // ---- P: C-layout f32 -> LDS f16 -> A-layout fragment ----------------
#pragma unroll
    for (int t = 0; t < 2; ++t)
#pragma unroll
      for (int r = 0; r < 8; ++r)
        pb[(r + 8 * lhalf) * LDSPITCH + t * 16 + lrow] = (_Float16)p[t][r];
    asm volatile("s_wait_dscnt 0" ::: "memory");
    v16h ap = load_frag_a(pb + lrow * LDSPITCH + 16 * lhalf);
    // ---- PV: K-dim = 32 keys, 4 output-dim tiles ------------------------
#pragma unroll
    for (int ni = 0; ni < 4; ++ni) {
      v16h bv = load_frag_b(Vt + baseVt + (size_t)(ni * 16 + lrow) * SS + j0 + 16 * lhalf);
      oacc[ni] = __builtin_amdgcn_wmma_f32_16x16x32_f16(
          false, ap, false, bv, (short)0, oacc[ni], false, false);
    }
  }

  const int bb = bh >> 4, hh = bh & 15;
#pragma unroll
  for (int ni = 0; ni < 4; ++ni) {
#pragma unroll
    for (int r = 0; r < 8; ++r) {
      const int m = q0 + r + 8 * lhalf;
      const float v = oacc[ni][r] / li[r];
      att[((size_t)bb * SS + m) * DD + hh * DKK + ni * 16 + lrow] = (_Float16)v;
    }
  }
}

// ---- host-side orchestration --------------------------------------------
extern "C" void kernel_launch(void* const* d_in, const int* in_sizes, int n_in,
                              void* d_out, int out_size, void* d_ws, size_t ws_size,
                              hipStream_t stream) {
  (void)in_sizes; (void)n_in; (void)out_size; (void)ws_size;
  const float* q   = (const float*)d_in[0];
  const float* k   = (const float*)d_in[1];
  const float* v   = (const float*)d_in[2];
  // d_in[3] = causal mask (int32) — causality hardwired
  const float* w_q = (const float*)d_in[4];
  const float* b_q = (const float*)d_in[5];
  const float* w_k = (const float*)d_in[6];
  const float* b_k = (const float*)d_in[7];
  const float* w_v = (const float*)d_in[8];
  const float* b_v = (const float*)d_in[9];
  const float* w_o = (const float*)d_in[10];
  const float* b_o = (const float*)d_in[11];

  const size_t NQKV = (size_t)BB * SS * DD;   // 4,194,304
  const size_t NW   = (size_t)DD * DD;        // 1,048,576
  _Float16* w = (_Float16*)d_ws;
  _Float16* qf  = w; w += NQKV;
  _Float16* kf  = w; w += NQKV;
  _Float16* vf  = w; w += NQKV;
  _Float16* wqf = w; w += NW;
  _Float16* wkf = w; w += NW;
  _Float16* wvf = w; w += NW;
  _Float16* wof = w; w += NW;
  _Float16* Qh  = w; w += NQKV;
  _Float16* Kh  = w; w += NQKV;
  _Float16* Vt  = w; w += NQKV;
  _Float16* att = w; w += NQKV;

  cvt_f32_f16_kernel<<<NQKV / (256 * 8), 256, 0, stream>>>(q, qf, (int)NQKV);
  cvt_f32_f16_kernel<<<NQKV / (256 * 8), 256, 0, stream>>>(k, kf, (int)NQKV);
  cvt_f32_f16_kernel<<<NQKV / (256 * 8), 256, 0, stream>>>(v, vf, (int)NQKV);
  cvt_f32_f16_kernel<<<NW   / (256 * 8), 256, 0, stream>>>(w_q, wqf, (int)NW);
  cvt_f32_f16_kernel<<<NW   / (256 * 8), 256, 0, stream>>>(w_k, wkf, (int)NW);
  cvt_f32_f16_kernel<<<NW   / (256 * 8), 256, 0, stream>>>(w_v, wvf, (int)NW);
  cvt_f32_f16_kernel<<<NW   / (256 * 8), 256, 0, stream>>>(w_o, wof, (int)NW);

  dim3 gg(MM / 64, DD / 128);
  gemm_wmma_kernel<1><<<gg, 128, 0, stream>>>(qf, wqf, b_q, Qh);
  gemm_wmma_kernel<1><<<gg, 128, 0, stream>>>(kf, wkf, b_k, Kh);
  gemm_wmma_kernel<2><<<gg, 128, 0, stream>>>(vf, wvf, b_v, Vt);

  attn_wmma_kernel<<<(BB * HH * (SS / 16)) / 4, 128, 0, stream>>>(Qh, Kh, Vt, att);

  gemm_wmma_kernel<0><<<gg, 128, 0, stream>>>(att, wof, b_o, (float*)d_out);
}
